// BiMamba_2791728742937
// MI455X (gfx1250) — compile-verified, matrix operand order checked by reading, not run on hardware
//
#include <hip/hip_runtime.h>

typedef __attribute__((ext_vector_type(16))) __bf16 bf16x16;
typedef __attribute__((ext_vector_type(8)))  __bf16 bf16x8;
typedef __attribute__((ext_vector_type(4)))  __bf16 bf16x4;
typedef __attribute__((ext_vector_type(8)))  float  f32x8;

#define LC    2048
#define ZSTR  2112   // 2096 padded to multiple of 64
#define CONVD 1056
#define DIN   1024
#define LROW  40               // LDS row stride in halves (pad vs bank conflicts)
#define BOFF  (128 * LROW)     // B tile offset (halves) inside a stage buffer

__device__ __forceinline__ int map_row_rev(int m) {
  return (m & ~(LC - 1)) + (LC - 1 - (m & (LC - 1)));
}

// ---- CDNA5 async global->LDS copy (16B per lane), tracked by ASYNCcnt ----
__device__ __forceinline__ void async_ld16(const void* gptr, unsigned lds_byte) {
  asm volatile("global_load_async_to_lds_b128 %0, %1, off"
               :: "v"(lds_byte), "v"((unsigned long long)(size_t)gptr)
               : "memory");
}
__device__ __forceinline__ void wait_async_le4() {
#if __has_builtin(__builtin_amdgcn_s_wait_asynccnt)
  __builtin_amdgcn_s_wait_asynccnt(4);
#else
  asm volatile("s_wait_asynccnt 0x4" ::: "memory");
#endif
}

// 16-bit WMMA fragment from row-major LDS row: lane = m + 16g,
// halves[0..7] <- K = 8g..8g+7 ; halves[8..15] <- K = 8g+16..8g+23.
__device__ __forceinline__ bf16x16 frag_from_lds(const __bf16* rowp, int g) {
  union { bf16x16 v; bf16x8 h[2]; } u;
  u.h[0] = *reinterpret_cast<const bf16x8*>(rowp + 8 * g);
  u.h[1] = *reinterpret_cast<const bf16x8*>(rowp + 8 * g + 16);
  return u.v;
}

// C[m,n] = sum_k A[m,k] * W[n,k]; A: MxK bf16, W: NxK bf16 (row-major).
// WG = 256 thr = 8 waves; wave tile 32x64 (2x4 WMMA 16x16x32); WG tile 128x128.
// Double-buffered LDS, filled by async global->LDS DMA.
__global__ __launch_bounds__(256) void gemm_bf16_wmma(
    const __bf16* __restrict__ A, int lda, int rev_in,
    const __bf16* __restrict__ W, int N, int K,
    float* __restrict__ C, int ldc, int rev_out, int accum, int n_lim)
{
  __shared__ __attribute__((aligned(16))) __bf16 sm[2][(128 + 128) * LROW];
  const int tid  = threadIdx.x;
  const int lane = tid & 31, wid = tid >> 5;
  const int wm = (wid & 3) * 32, wn = (wid >> 2) * 64;
  const int m0 = blockIdx.y * 128, n0 = blockIdx.x * 128;

  const unsigned lb0 = (unsigned)(size_t)&sm[0][0];
  const unsigned lb1 = (unsigned)(size_t)&sm[1][0];

  f32x8 acc[2][4];
#pragma unroll
  for (int i = 0; i < 2; ++i)
#pragma unroll
    for (int j = 0; j < 4; ++j) acc[i][j] = (f32x8){};

  const int nk = K >> 5;

  // ---- stage loader: A tile 128x32 + B tile 128x32 (4 async instrs/thread) ----
  auto stage_load = [&](int k0, unsigned lbase) {
#pragma unroll
    for (int i = 0; i < 2; ++i) {                       // A: 512 x 16B segments
      int lin = tid + 256 * i;
      int row = lin >> 2, seg = lin & 3;
      int gm = m0 + row;
      if (rev_in) gm = map_row_rev(gm);
      async_ld16(&A[(size_t)gm * lda + k0 + seg * 8],
                 lbase + (unsigned)(row * LROW + seg * 8) * 2);
    }
#pragma unroll
    for (int i = 0; i < 2; ++i) {                       // B: 512 x 16B segments
      int lin = tid + 256 * i;
      int row = lin >> 2, seg = lin & 3;
      int gn = n0 + row;
      if (gn >= N) gn = 0;                              // clamp (cols masked on store)
      async_ld16(&W[(size_t)gn * K + k0 + seg * 8],
                 lbase + (unsigned)(BOFF + row * LROW + seg * 8) * 2);
    }
  };

  stage_load(0, lb0);                                   // prologue

  for (int kk = 0; kk < nk; ++kk) {
    const int knext = (kk + 1 < nk) ? (kk + 1) : 0;     // last prefetch is redundant
    stage_load(knext << 5, (kk & 1) ? lb0 : lb1);
    wait_async_le4();                                   // stage kk resident
    __syncthreads();

    const __bf16* base = &sm[kk & 1][0];
    const int ml = lane & 15, g = lane >> 4;
    bf16x16 a0 = frag_from_lds(base + (wm + ml) * LROW, g);
    bf16x16 a1 = frag_from_lds(base + (wm + 16 + ml) * LROW, g);
    bf16x16 b0 = frag_from_lds(base + BOFF + (wn + ml) * LROW, g);
    bf16x16 b1 = frag_from_lds(base + BOFF + (wn + 16 + ml) * LROW, g);
    bf16x16 b2 = frag_from_lds(base + BOFF + (wn + 32 + ml) * LROW, g);
    bf16x16 b3 = frag_from_lds(base + BOFF + (wn + 48 + ml) * LROW, g);
    acc[0][0] = __builtin_amdgcn_wmma_f32_16x16x32_bf16(false, a0, false, b0, (short)0, acc[0][0], false, false);
    acc[0][1] = __builtin_amdgcn_wmma_f32_16x16x32_bf16(false, a0, false, b1, (short)0, acc[0][1], false, false);
    acc[0][2] = __builtin_amdgcn_wmma_f32_16x16x32_bf16(false, a0, false, b2, (short)0, acc[0][2], false, false);
    acc[0][3] = __builtin_amdgcn_wmma_f32_16x16x32_bf16(false, a0, false, b3, (short)0, acc[0][3], false, false);
    acc[1][0] = __builtin_amdgcn_wmma_f32_16x16x32_bf16(false, a1, false, b0, (short)0, acc[1][0], false, false);
    acc[1][1] = __builtin_amdgcn_wmma_f32_16x16x32_bf16(false, a1, false, b1, (short)0, acc[1][1], false, false);
    acc[1][2] = __builtin_amdgcn_wmma_f32_16x16x32_bf16(false, a1, false, b2, (short)0, acc[1][2], false, false);
    acc[1][3] = __builtin_amdgcn_wmma_f32_16x16x32_bf16(false, a1, false, b3, (short)0, acc[1][3], false, false);
    __syncthreads();                                    // stage kk free for reuse
  }

  // ---- store: VGPR r -> M = r + 8*(lane>=16), N = lane%16 ----
  const int nl = lane & 15, half8 = (lane >> 4) * 8;
#pragma unroll
  for (int i = 0; i < 2; ++i) {
#pragma unroll
    for (int j = 0; j < 4; ++j) {
      int n = n0 + wn + j * 16 + nl;
      if (n < n_lim) {
#pragma unroll
        for (int r = 0; r < 8; ++r) {
          int m = m0 + wm + i * 16 + half8 + r;
          int gm = rev_out ? map_row_rev(m) : m;
          size_t off = (size_t)gm * ldc + n;
          if (accum) C[off] += acc[i][j][r]; else C[off] = acc[i][j][r];
        }
      }
    }
  }
}

// fp32 -> bf16 bulk convert (vector of 4)
__global__ __launch_bounds__(256) void cvt_f32_bf16(
    const float* __restrict__ s, __bf16* __restrict__ d, int n)
{
  int i = (blockIdx.x * 256 + threadIdx.x) * 4;
  if (i >= n) return;
  const float4 v = *reinterpret_cast<const float4*>(s + i);
  bf16x4 b;
  b[0] = (__bf16)v.x; b[1] = (__bf16)v.y; b[2] = (__bf16)v.z; b[3] = (__bf16)v.w;
  *reinterpret_cast<bf16x4*>(d + i) = b;
}

// causal depthwise conv1d (width 4) + bias + silu over xBC slice of zxbcdt
__global__ __launch_bounds__(256) void conv_silu(
    const float* __restrict__ zx, const float* __restrict__ cw,
    const float* __restrict__ cb, float* __restrict__ xbc)
{
  int idx = blockIdx.x * 256 + threadIdx.x;
  if (idx >= 8 * LC * CONVD) return;
  int c  = idx % CONVD;
  int bl = idx / CONVD;
  int l  = bl & (LC - 1);
  float acc = cb[c];
  const float* w = cw + c * 4;
#pragma unroll
  for (int j = 0; j < 4; ++j) {
    int ls = l - 3 + j;
    if (ls >= 0) acc += w[j] * zx[(size_t)(bl - 3 + j) * ZSTR + DIN + c];
  }
  xbc[(size_t)bl * CONVD + c] = acc * (1.f / (1.f + __expf(-acc)));
}

// selective-state scan: one block per (head, batch); thread p owns 16-state row
__global__ __launch_bounds__(64) void scan_kernel(
    const float* __restrict__ zx, const float* __restrict__ xbc,
    const float* __restrict__ dt_bias, const float* __restrict__ A_log,
    const float* __restrict__ Dp, float* __restrict__ ybuf)
{
  const int h = blockIdx.x, b = blockIdx.y, p = threadIdx.x;
  __shared__ float xs[64][64];
  __shared__ float Bc[64][16], Cc[64][16], dAs[64], dts[64];

  float hst[16];
#pragma unroll
  for (int n = 0; n < 16; ++n) hst[n] = 0.f;
  const float dtb  = dt_bias[h];
  const float Aneg = -__expf(A_log[h]);
  const float dp   = Dp[h];

  for (int t0 = 0; t0 < LC; t0 += 64) {
    for (int tt = 0; tt < 64; ++tt)
      xs[tt][p] = xbc[(size_t)(b * LC + t0 + tt) * CONVD + h * 64 + p];
    if (p < 16)
      for (int tt = 0; tt < 64; ++tt)
        Bc[tt][p] = xbc[(size_t)(b * LC + t0 + tt) * CONVD + DIN + p];
    else if (p < 32)
      for (int tt = 0; tt < 64; ++tt)
        Cc[tt][p - 16] = xbc[(size_t)(b * LC + t0 + tt) * CONVD + DIN + 16 + (p - 16)];
    {
      int tt = p;
      float v  = zx[(size_t)(b * LC + t0 + tt) * ZSTR + 2080 + h] + dtb;
      float dt = (v > 20.f) ? v : __logf(1.f + __expf(v));   // softplus
      dts[tt] = dt;
      dAs[tt] = __expf(dt * Aneg);
    }
    __syncthreads();

    for (int tt = 0; tt < 64; ++tt) {
      float dA = dAs[tt];
      float xv = xs[tt][p];
      float s  = dts[tt] * xv;
      float yv = 0.f;
#pragma unroll
      for (int n = 0; n < 16; ++n) {
        hst[n] = hst[n] * dA + s * Bc[tt][n];
        yv += hst[n] * Cc[tt][n];
      }
      ybuf[(size_t)(b * LC + t0 + tt) * DIN + h * 64 + p] = yv + dp * xv;
    }
    __syncthreads();
  }
}

// ybf = bf16( rmsnorm(y * silu(z)) * norm_w ); z = zxbcdt[:, :DIN]
__global__ __launch_bounds__(256) void gated_rmsnorm(
    const float* __restrict__ ybuf, const float* __restrict__ zx,
    const float* __restrict__ nw, __bf16* __restrict__ ybf)
{
  const int row = blockIdx.x, t = threadIdx.x;
  __shared__ float red[256];
  const size_t yb = (size_t)row * DIN;
  const size_t zb = (size_t)row * ZSTR;
  float g[4], ss = 0.f;
#pragma unroll
  for (int i = 0; i < 4; ++i) {
    int c = t + 256 * i;
    float z = zx[zb + c];
    float y = ybuf[yb + c];
    float gv = y * (z / (1.f + __expf(-z)));
    g[i] = gv; ss += gv * gv;
  }
  red[t] = ss; __syncthreads();
  for (int s = 128; s > 0; s >>= 1) {
    if (t < s) red[t] += red[t + s];
    __syncthreads();
  }
  float scale = rsqrtf(red[0] * (1.f / (float)DIN) + 1e-5f);
#pragma unroll
  for (int i = 0; i < 4; ++i) {
    int c = t + 256 * i;
    ybf[yb + c] = (__bf16)(g[i] * scale * nw[c]);
  }
}

extern "C" void kernel_launch(void* const* d_in, const int* in_sizes, int n_in,
                              void* d_out, int out_size, void* d_ws, size_t ws_size,
                              hipStream_t stream) {
  (void)in_sizes; (void)n_in; (void)out_size; (void)ws_size;
  const int Bb = 8, M = Bb * LC;  // 16384 rows
  const float* x   = (const float*)d_in[0];
  float* out = (float*)d_out;

  float*  zx   = (float*)d_ws;                      // M x 2112 f32
  float*  xbc  = zx  + (size_t)M * ZSTR;            // M x 1056 f32
  float*  ybuf = xbc + (size_t)M * CONVD;           // M x 1024 f32
  __bf16* xbf  = (__bf16*)(ybuf + (size_t)M * DIN); // M x 512  bf16
  __bf16* ybf  = xbf + (size_t)M * 512;             // M x 1024 bf16
  __bf16* wbf  = ybf + (size_t)M * DIN;             // up to 2096x512 bf16

  // x -> bf16 once (shared by both directions)
  const int nx = M * 512;
  cvt_f32_bf16<<<(nx / 4 + 255) / 256, 256, 0, stream>>>(x, xbf, nx);

  for (int dir = 0; dir < 2; ++dir) {
    const float* inW  = (const float*)d_in[1 + dir * 8 + 0];
    const float* cw   = (const float*)d_in[1 + dir * 8 + 1];
    const float* cb   = (const float*)d_in[1 + dir * 8 + 2];
    const float* dtb  = (const float*)d_in[1 + dir * 8 + 3];
    const float* alog = (const float*)d_in[1 + dir * 8 + 4];
    const float* dp   = (const float*)d_in[1 + dir * 8 + 5];
    const float* nw   = (const float*)d_in[1 + dir * 8 + 6];
    const float* outW = (const float*)d_in[1 + dir * 8 + 7];

    // in_proj: zx = x @ inW^T (bwd reads rows reversed)
    const int nw1 = 2096 * 512;
    cvt_f32_bf16<<<(nw1 / 4 + 255) / 256, 256, 0, stream>>>(inW, wbf, nw1);
    dim3 g1((2096 + 127) / 128, M / 128);
    gemm_bf16_wmma<<<g1, 256, 0, stream>>>(xbf, 512, dir, wbf, 2096, 512,
                                           zx, ZSTR, 0, 0, ZSTR);

    const int convN = M * CONVD;
    conv_silu<<<(convN + 255) / 256, 256, 0, stream>>>(zx, cw, cb, xbc);
    scan_kernel<<<dim3(16, Bb), 64, 0, stream>>>(zx, xbc, dtb, alog, dp, ybuf);
    gated_rmsnorm<<<M, 256, 0, stream>>>(ybuf, zx, nw, ybf);

    // out_proj: out (+)= yn @ outW^T (bwd writes rows reversed, accumulates)
    const int nw2 = 512 * 1024;
    cvt_f32_bf16<<<(nw2 / 4 + 255) / 256, 256, 0, stream>>>(outW, wbf, nw2);
    dim3 g2(512 / 128, M / 128);
    gemm_bf16_wmma<<<g2, 256, 0, stream>>>(ybf, 1024, 0, wbf, 512, 1024,
                                           out, 512, dir, dir, 512);
  }
}
